// MaskedMultiHeadAttention_62105227100866
// MI455X (gfx1250) — compile-verified, matrix operand order checked by reading, not run on hardware
//
#include <hip/hip_runtime.h>
#include <hip/hip_bf16.h>

typedef unsigned short ushort_t;
typedef __attribute__((ext_vector_type(16))) __bf16 v16bf;
typedef __attribute__((ext_vector_type(8)))  float  v8f;

constexpr int cS  = 1024;
constexpr int cP  = 1024;
constexpr int cB  = 4;
constexpr int cE  = 1024;
constexpr int cH  = 16;
constexpr int cI  = 64;
constexpr int cJ  = cP + cS;      // 2048
constexpr int cHI = cH * cI;      // 1024
constexpr int c2HI = 2 * cHI;     // 2048
constexpr float cSCALE = 0.125f;  // 1/sqrt(64)
constexpr float cLOG2E = 1.4426950408889634f;

#define WMMA_BF16(a, b, c) \
  __builtin_amdgcn_wmma_f32_16x16x32_bf16(false, (a), false, (b), (short)0, (c), false, false)

__device__ __forceinline__ ushort_t f2bf(float f) {
  unsigned u = __float_as_uint(f);
  u += 0x7FFFu + ((u >> 16) & 1u);
  return (ushort_t)(u >> 16);
}

// Load a 16x32 bf16 tile (A-operand layout; B-operand is the mirror with N
// striped across lanes, same load pattern from a K-major (row, k) array).
// Per ISA 7.12.2: lane m (0-15): K 0..7 then 16..23; lane m+16: K 8..15 then 24..31.
__device__ __forceinline__ v16bf load_t(const ushort_t* __restrict__ p, int ld) {
  int lane = threadIdx.x & 31;
  int row = lane & 15, hi = lane >> 4;
  const ushort_t* q = p + (size_t)row * ld + hi * 8;
  union { v16bf v; uint4 u[2]; } r;
  r.u[0] = *(const uint4*)q;
  r.u[1] = *(const uint4*)(q + 16);
  return r.v;
}

__device__ __forceinline__ v16bf load_t_clamp(const ushort_t* __restrict__ base, int ld,
                                              int row0, int rowmax, int colofs) {
  int lane = threadIdx.x & 31;
  int rr = row0 + (lane & 15);
  if (rr > rowmax) rr = rowmax;
  int hi = lane >> 4;
  const ushort_t* q = base + (size_t)rr * ld + colofs + hi * 8;
  union { v16bf v; uint4 u[2]; } r;
  r.u[0] = *(const uint4*)q;
  r.u[1] = *(const uint4*)(q + 16);
  return r.v;
}

// ---------------------------------------------------------------- converts
__global__ void cvt_f32_bf16(const float* __restrict__ src, ushort_t* __restrict__ dst, long n) {
  long i = (long)blockIdx.x * blockDim.x + threadIdx.x;
  if (i < n) dst[i] = f2bf(src[i]);
}

// ---------------------------------------------------------------- LDS-blocked GEMM
// C[M,N](fp32) = A[M,K](bf16) @ W[N,K](bf16)^T (+res).
// Workgroup tile 128x128: 8 waves (4x2), 32x64 per wave, k-sliced by 32 via LDS.
// ~120 VGPRs/wave: 64 accum + 16 A-ops + 8 B-op + 16 pipeline regs (no spills).
constexpr int GEMM_LDA = 40;  // 32 cols + 8 ushort pad (keeps 16B align, spreads banks)

__launch_bounds__(256)
__global__ void gemm_bf16_wmma(const ushort_t* __restrict__ A, const ushort_t* __restrict__ W,
                               float* __restrict__ C, const float* __restrict__ res,
                               int M, int N, int K) {
  __shared__ alignas(16) ushort_t As[128 * GEMM_LDA];
  __shared__ alignas(16) ushort_t Bs[128 * GEMM_LDA];

  int Nt = N >> 7;
  int m0 = (blockIdx.x / Nt) << 7;
  int n0 = (blockIdx.x % Nt) << 7;
  int tid = threadIdx.x;
  int wave = tid >> 5, lane = tid & 31;
  int wm = (wave >> 1) * 32;   // 4 row groups of 32
  int wn = (wave & 1) * 64;    // 2 col groups of 64
  int rowsel = lane & 15, hi = lane >> 4;

  // staging: 2 threads per row, 16 cols (32B) each
  int ldrow = tid >> 1, ldhalf = (tid & 1) * 16;
  const ushort_t* Ap = A + (size_t)(m0 + ldrow) * K + ldhalf;
  const ushort_t* Bp = W + (size_t)(n0 + ldrow) * K + ldhalf;
  ushort_t* Asd = &As[ldrow * GEMM_LDA + ldhalf];
  ushort_t* Bsd = &Bs[ldrow * GEMM_LDA + ldhalf];

  v8f acc[8];
#pragma unroll
  for (int i = 0; i < 8; i++) acc[i] = {};

  uint4 ga0 = *(const uint4*)(Ap);
  uint4 ga1 = *(const uint4*)(Ap + 8);
  uint4 gb0 = *(const uint4*)(Bp);
  uint4 gb1 = *(const uint4*)(Bp + 8);

  for (int kc = 0; kc < K; kc += 32) {
    // stage current k-slice to LDS
    *(uint4*)Asd = ga0;
    *(uint4*)(Asd + 8) = ga1;
    *(uint4*)Bsd = gb0;
    *(uint4*)(Bsd + 8) = gb1;
    __syncthreads();
    // prefetch next k-slice into registers while computing from LDS
    if (kc + 32 < K) {
      ga0 = *(const uint4*)(Ap + kc + 32);
      ga1 = *(const uint4*)(Ap + kc + 40);
      gb0 = *(const uint4*)(Bp + kc + 32);
      gb1 = *(const uint4*)(Bp + kc + 40);
    }
    // 32x64 per wave = 2x4 subtiles, operands from LDS
    v16bf aop[2];
#pragma unroll
    for (int mi = 0; mi < 2; mi++) {
      int r = wm + mi * 16 + rowsel;
      union { v16bf v; uint4 u[2]; } t;
      t.u[0] = *(const uint4*)&As[r * GEMM_LDA + hi * 8];
      t.u[1] = *(const uint4*)&As[r * GEMM_LDA + hi * 8 + 16];
      aop[mi] = t.v;
    }
#pragma unroll
    for (int ni = 0; ni < 4; ni++) {
      int r = wn + ni * 16 + rowsel;
      union { v16bf v; uint4 u[2]; } t;
      t.u[0] = *(const uint4*)&Bs[r * GEMM_LDA + hi * 8];
      t.u[1] = *(const uint4*)&Bs[r * GEMM_LDA + hi * 8 + 16];
      v16bf bop = t.v;
      acc[ni] = WMMA_BF16(aop[0], bop, acc[ni]);
      acc[4 + ni] = WMMA_BF16(aop[1], bop, acc[4 + ni]);
    }
    __syncthreads();
  }

#pragma unroll
  for (int mi = 0; mi < 2; mi++)
#pragma unroll
    for (int ni = 0; ni < 4; ni++) {
      v8f a = acc[mi * 4 + ni];
#pragma unroll
      for (int v = 0; v < 8; v++) {
        size_t row = m0 + wm + mi * 16 + 8 * hi + v;
        size_t col = n0 + wn + ni * 16 + rowsel;
        size_t idx = row * N + col;
        float val = a[v];
        if (res) val += res[idx];
        C[idx] = val;
      }
    }
}

// ---------------------------------------------------------------- repacks
__global__ void repack_kv(const float* __restrict__ C, ushort_t* __restrict__ Kb,
                          ushort_t* __restrict__ Vt) {
  long idx = (long)blockIdx.x * blockDim.x + threadIdx.x;
  if (idx >= (long)cB * cH * cJ * cI) return;
  int i = idx % cI; long t = idx / cI;
  int j = t % cJ; t /= cJ;
  int h = t % cH; int b = (int)(t / cH);
  size_t crow = (size_t)j * cB + b;
  Kb[idx] = f2bf(C[crow * c2HI + h * cI + i]);
  Vt[((size_t)(b * cH + h) * cI + i) * cJ + j] = f2bf(C[crow * c2HI + cHI + h * cI + i]);
}

__global__ void repack_q(const float* __restrict__ C, const float* __restrict__ u,
                         const float* __restrict__ v, ushort_t* __restrict__ qu,
                         ushort_t* __restrict__ qv) {
  long idx = (long)blockIdx.x * blockDim.x + threadIdx.x;
  if (idx >= (long)cB * cH * cS * cI) return;
  int i = idx % cI; long t = idx / cI;
  int s = t % cS; t /= cS;
  int h = t % cH; int b = (int)(t / cH);
  float qval = C[((size_t)s * cB + b) * cHI + h * cI + i];
  qu[idx] = f2bf(qval + u[h * cI + i]);
  qv[idx] = f2bf(qval + v[h * cI + i]);
}

__global__ void repack_p(const float* __restrict__ C, ushort_t* __restrict__ Pv) {
  long idx = (long)blockIdx.x * blockDim.x + threadIdx.x;
  if (idx >= (long)cH * cJ * cI) return;
  int i = idx % cI; long t = idx / cI;
  int j = t % cJ; int h = (int)(t / cJ);
  Pv[idx] = f2bf(C[(size_t)j * cHI + h * cI + i]);
}

// ---------------------------------------------------------------- fused flash attention
// 1 wave per (b,h, 16-row query tile). Key loop in tiles of 32 columns.
// rel_shift(pos)[s,j] = pos[s, j + S-1-s]  (valid region only; rest masked).
__launch_bounds__(32)
__global__ void attn_flash(const ushort_t* __restrict__ qu, const ushort_t* __restrict__ qv,
                           const ushort_t* __restrict__ Kb, const ushort_t* __restrict__ Vt,
                           const ushort_t* __restrict__ Pv, ushort_t* __restrict__ ctx) {
  __shared__ alignas(16) float posLDS[16 * 48];
  __shared__ alignas(16) float scoLDS[16 * 32];
  __shared__ alignas(16) ushort_t prbLDS[16 * 32];
  __shared__ float alphaLDS[16];
  __shared__ float lrowLDS[16];
  __shared__ float red32[32];

  int tilesPerBH = cS / 16;
  int bh = blockIdx.x / tilesPerBH;
  int s0 = (blockIdx.x % tilesPerBH) * 16;
  int h = bh % cH;
  int b = bh / cH;
  int lane = threadIdx.x & 31;
  int rowid = lane & 15, hi = lane >> 4;

  // Preload query tiles (A operands), two 32-wide K-chunks over I=64
  const ushort_t* quB = qu + ((size_t)bh * cS + s0) * cI;
  const ushort_t* qvB = qv + ((size_t)bh * cS + s0) * cI;
  v16bf quA[2], qvA[2];
#pragma unroll
  for (int kc = 0; kc < 2; kc++) {
    quA[kc] = load_t(quB + kc * 32, cI);
    qvA[kc] = load_t(qvB + kc * 32, cI);
  }

  v8f ctxAcc[4];
  ctxAcc[0] = {}; ctxAcc[1] = {}; ctxAcc[2] = {}; ctxAcc[3] = {};
  float mrow = -1e30f, lsum = 0.f;

  const ushort_t* Pvh = Pv + (size_t)h * cJ * cI;
  int nt = (s0 + 15 + cP) / 32 + 1;

  for (int t = 0; t < nt; t++) {
    int j0 = t * 32;
    // ---- content scores: (q+u) @ K^T, 16x32 output via 2 n-subtiles
    v8f c0 = {}, c1 = {};
    const ushort_t* Kt = Kb + ((size_t)bh * cJ + j0) * cI;
#pragma unroll
    for (int kc = 0; kc < 2; kc++) {
      c0 = WMMA_BF16(quA[kc], load_t(Kt + kc * 32, cI), c0);
      c1 = WMMA_BF16(quA[kc], load_t(Kt + (size_t)16 * cI + kc * 32, cI), c1);
    }
    // ---- positional scores: (q+v) @ Pv^T over k band [k0, k0+47]
    int k0 = j0 + cS - 16 - s0;  // 16-aligned, >= 0
    v8f p0 = {}, p1 = {}, p2 = {};
#pragma unroll
    for (int kc = 0; kc < 2; kc++) {
      p0 = WMMA_BF16(qvA[kc], load_t_clamp(Pvh, cI, k0, cJ - 1, kc * 32), p0);
      p1 = WMMA_BF16(qvA[kc], load_t_clamp(Pvh, cI, k0 + 16, cJ - 1, kc * 32), p1);
      p2 = WMMA_BF16(qvA[kc], load_t_clamp(Pvh, cI, k0 + 32, cJ - 1, kc * 32), p2);
    }
    // ---- dump to LDS (C layout: row = v + 8*hi, col = lane&15)
#pragma unroll
    for (int v = 0; v < 8; v++) {
      int rr = v + 8 * hi;
      scoLDS[rr * 32 + rowid] = c0[v];
      scoLDS[rr * 32 + 16 + rowid] = c1[v];
      posLDS[rr * 48 + rowid] = p0[v];
      posLDS[rr * 48 + 16 + rowid] = p1[v];
      posLDS[rr * 48 + 32 + rowid] = p2[v];
    }
    __syncthreads();

    // ---- softmax: lane owns half-row (row=rowid, cols hi*16..hi*16+15)
    float sc[16];
    float tmax = -1e30f;
    int limit = s0 + rowid + cP - j0;  // valid iff col <= limit
#pragma unroll
    for (int c = 0; c < 16; c++) {
      int cc = hi * 16 + c;
      float vv = (scoLDS[rowid * 32 + cc] + posLDS[rowid * 48 + cc + 15 - rowid]) * cSCALE;
      if (cc > limit) vv = -1e30f;
      sc[c] = vv;
      tmax = fmaxf(tmax, vv);
    }
    red32[lane] = tmax;
    __syncthreads();
    tmax = fmaxf(red32[lane], red32[lane ^ 16]);
    float mnew = fmaxf(mrow, tmax);
    float al = exp2f((mrow - mnew) * cLOG2E);
    float ps = 0.f;
#pragma unroll
    for (int c = 0; c < 16; c++) {
      float pr = exp2f((sc[c] - mnew) * cLOG2E);
      ps += pr;
      prbLDS[rowid * 32 + hi * 16 + c] = f2bf(pr);
    }
    red32[lane] = ps;
    __syncthreads();
    ps = red32[lane] + red32[lane ^ 16];
    lsum = lsum * al + ps;
    mrow = mnew;
    if (hi == 0) alphaLDS[rowid] = al;
    __syncthreads();

    // ---- rescale running context
    float alv[8];
#pragma unroll
    for (int v = 0; v < 8; v++) alv[v] = alphaLDS[v + 8 * hi];
#pragma unroll
    for (int s2 = 0; s2 < 4; s2++)
#pragma unroll
      for (int v = 0; v < 8; v++) ctxAcc[s2][v] *= alv[v];

    // ---- probs A-operand from LDS (A layout: lane m: cols hi*8..+7, hi*8+16..+23)
    union { v16bf v; uint4 u[2]; } pa;
    pa.u[0] = *(const uint4*)&prbLDS[rowid * 32 + hi * 8];
    pa.u[1] = *(const uint4*)&prbLDS[rowid * 32 + hi * 8 + 16];

    // ---- ctx += P(16x32) @ V(32x64); Vt is (b,h, i, j) so B-operand rows are contiguous j
#pragma unroll
    for (int sub = 0; sub < 4; sub++) {
      const ushort_t* Vb = Vt + ((size_t)bh * cI + sub * 16) * cJ + j0;
      ctxAcc[sub] = WMMA_BF16(pa.v, load_t(Vb, cJ), ctxAcc[sub]);
    }
    __syncthreads();
  }

  // ---- finalize: divide by row sums, store bf16 ctx as (s, b, h*I + i)
  if (hi == 0) lrowLDS[rowid] = lsum;
  __syncthreads();
  float linv[8];
#pragma unroll
  for (int v = 0; v < 8; v++) linv[v] = 1.f / lrowLDS[v + 8 * hi];
#pragma unroll
  for (int sub = 0; sub < 4; sub++)
#pragma unroll
    for (int v = 0; v < 8; v++) {
      int rr = v + 8 * hi;
      float val = ctxAcc[sub][v] * linv[v];
      ctx[((size_t)(s0 + rr) * cB + b) * cHI + h * cI + sub * 16 + rowid] = f2bf(val);
    }
}

// ---------------------------------------------------------------- LayerNorm
__launch_bounds__(256)
__global__ void ln_kernel(const float* __restrict__ X, const float* __restrict__ w,
                          const float* __restrict__ bb, float* __restrict__ out) {
  __shared__ float red[256];
  size_t row = blockIdx.x;
  const float* x = X + row * cE;
  float s = 0.f, s2 = 0.f;
  for (int c = threadIdx.x; c < cE; c += 256) {
    float v = x[c];
    s += v;
    s2 += v * v;
  }
  red[threadIdx.x] = s;
  __syncthreads();
  for (int st = 128; st > 0; st >>= 1) {
    if (threadIdx.x < st) red[threadIdx.x] += red[threadIdx.x + st];
    __syncthreads();
  }
  float mean = red[0] / cE;
  __syncthreads();
  red[threadIdx.x] = s2;
  __syncthreads();
  for (int st = 128; st > 0; st >>= 1) {
    if (threadIdx.x < st) red[threadIdx.x] += red[threadIdx.x + st];
    __syncthreads();
  }
  float var = red[0] / cE - mean * mean;
  float inv = rsqrtf(var + 1e-5f);
  for (int c = threadIdx.x; c < cE; c += 256)
    out[row * cE + c] = (x[c] - mean) * inv * w[c] + bb[c];
}

// ---------------------------------------------------------------- launch
extern "C" void kernel_launch(void* const* d_in, const int* in_sizes, int n_in,
                              void* d_out, int out_size, void* d_ws, size_t ws_size,
                              hipStream_t stream) {
  (void)in_sizes; (void)n_in; (void)out_size; (void)ws_size;
  const float* inp  = (const float*)d_in[0];
  const float* posE = (const float*)d_in[1];
  const float* mem  = (const float*)d_in[2];
  const float* u_   = (const float*)d_in[3];
  const float* v_   = (const float*)d_in[4];
  const float* Wkv  = (const float*)d_in[5];
  const float* Wq   = (const float*)d_in[6];
  const float* Wp   = (const float*)d_in[7];
  const float* Wout = (const float*)d_in[8];
  const float* lnw  = (const float*)d_in[9];
  const float* lnb  = (const float*)d_in[10];
  float* out = (float*)d_out;

  char* ws = (char*)d_ws;
  size_t off = 0;
  auto alloc = [&](size_t bytes) -> void* {
    void* p = ws + off;
    off += (bytes + 255) & ~(size_t)255;
    return p;
  };
  ushort_t* xmem_bf = (ushort_t*)alloc((size_t)cJ * cB * cE * 2);
  ushort_t* Wkv_bf  = (ushort_t*)alloc((size_t)c2HI * cE * 2);
  ushort_t* Wq_bf   = (ushort_t*)alloc((size_t)cHI * cE * 2);
  ushort_t* Wp_bf   = (ushort_t*)alloc((size_t)cHI * cE * 2);
  ushort_t* Wout_bf = (ushort_t*)alloc((size_t)cE * cHI * 2);
  ushort_t* pos_bf  = (ushort_t*)alloc((size_t)cJ * cE * 2);
  float*    Cbig    = (float*)alloc((size_t)(cJ * cB) * c2HI * 4);  // reused across stages
  ushort_t* Kbf     = (ushort_t*)alloc((size_t)cB * cH * cJ * cI * 2);
  ushort_t* Vtbf    = (ushort_t*)alloc((size_t)cB * cH * cI * cJ * 2);
  ushort_t* qu_bf   = (ushort_t*)alloc((size_t)cB * cH * cS * cI * 2);
  ushort_t* qv_bf   = (ushort_t*)alloc((size_t)cB * cH * cS * cI * 2);
  ushort_t* Pv_bf   = (ushort_t*)alloc((size_t)cH * cJ * cI * 2);
  ushort_t* ctx_bf  = (ushort_t*)alloc((size_t)cS * cB * cHI * 2);

  auto cdiv = [](long a, long b) { return (int)((a + b - 1) / b); };

  // Stage 0: fp32 -> bf16
  cvt_f32_bf16<<<cdiv((long)cP * cB * cE, 256), 256, 0, stream>>>(mem, xmem_bf, (long)cP * cB * cE);
  cvt_f32_bf16<<<cdiv((long)cS * cB * cE, 256), 256, 0, stream>>>(
      inp, xmem_bf + (size_t)cP * cB * cE, (long)cS * cB * cE);
  cvt_f32_bf16<<<cdiv((long)c2HI * cE, 256), 256, 0, stream>>>(Wkv, Wkv_bf, (long)c2HI * cE);
  cvt_f32_bf16<<<cdiv((long)cHI * cE, 256), 256, 0, stream>>>(Wq, Wq_bf, (long)cHI * cE);
  cvt_f32_bf16<<<cdiv((long)cHI * cE, 256), 256, 0, stream>>>(Wp, Wp_bf, (long)cHI * cE);
  cvt_f32_bf16<<<cdiv((long)cE * cHI, 256), 256, 0, stream>>>(Wout, Wout_bf, (long)cE * cHI);
  cvt_f32_bf16<<<cdiv((long)cJ * cE, 256), 256, 0, stream>>>(posE, pos_bf, (long)cJ * cE);

  // Stage 1: kv projection  (8192 x 2048 x 1024)
  {
    int M = cJ * cB, N = c2HI;
    gemm_bf16_wmma<<<(M / 128) * (N / 128), 256, 0, stream>>>(xmem_bf, Wkv_bf, Cbig, nullptr, M, N, cE);
    repack_kv<<<cdiv((long)cB * cH * cJ * cI, 256), 256, 0, stream>>>(Cbig, Kbf, Vtbf);
  }
  // Stage 2: q projection (4096 x 1024 x 1024), + u / + v
  {
    int M = cS * cB, N = cHI;
    gemm_bf16_wmma<<<(M / 128) * (N / 128), 256, 0, stream>>>(
        xmem_bf + (size_t)cP * cB * cE, Wq_bf, Cbig, nullptr, M, N, cE);
    repack_q<<<cdiv((long)cB * cH * cS * cI, 256), 256, 0, stream>>>(Cbig, u_, v_, qu_bf, qv_bf);
  }
  // Stage 3: positional projection (2048 x 1024 x 1024)
  {
    int M = cJ, N = cHI;
    gemm_bf16_wmma<<<(M / 128) * (N / 128), 256, 0, stream>>>(pos_bf, Wp_bf, Cbig, nullptr, M, N, cE);
    repack_p<<<cdiv((long)cH * cJ * cI, 256), 256, 0, stream>>>(Cbig, Pv_bf);
  }
  // Stage 4: fused rel-shift flash attention
  attn_flash<<<cB * cH * (cS / 16), 32, 0, stream>>>(qu_bf, qv_bf, Kbf, Vtbf, Pv_bf, ctx_bf);

  // Stage 5: output projection + residual (4096 x 1024 x 1024)
  {
    int M = cS * cB, N = cE;
    gemm_bf16_wmma<<<(M / 128) * (N / 128), 256, 0, stream>>>(ctx_bf, Wout_bf, Cbig, inp, M, N, cHI);
  }
  // Stage 6: LayerNorm
  ln_kernel<<<cS * cB, 256, 0, stream>>>(Cbig, lnw, lnb, out);
}
